// RNN_24910810317048
// MI455X (gfx1250) — matrix operand order repaired and checked by reading, not compiled, not run on hardware
//
#include <hip/hip_runtime.h>
#include <hip/hip_bf16.h>
#include <math.h>

typedef __bf16 v8bf  __attribute__((ext_vector_type(8)));
typedef __bf16 v16bf __attribute__((ext_vector_type(16)));
typedef float  v8f   __attribute__((ext_vector_type(8)));
typedef float  f32x4 __attribute__((ext_vector_type(4)));
typedef int    i32x4v __attribute__((vector_size(16)));   // matches builtin param type

#define NB 128      // batch
#define TT 512      // timesteps
#define DD 1024     // input dim
#define HH 1024     // hidden dim
#define NBLK2 16    // persistent blocks in phase 2

#if __has_builtin(__builtin_amdgcn_global_load_async_to_lds_b128) && \
    __has_builtin(__builtin_amdgcn_s_wait_asynccnt)
#define USE_ASYNC_LDS 1
#else
#define USE_ASYNC_LDS 0
#endif

#if USE_ASYNC_LDS
typedef __attribute__((address_space(1))) i32x4v* gas1_b128;
typedef __attribute__((address_space(3))) i32x4v* las3_b128;
#endif

static __device__ __forceinline__ v16bf cat16(v8bf a, v8bf b) {
  return __builtin_shufflevector(a, b, 0,1,2,3,4,5,6,7,8,9,10,11,12,13,14,15);
}
static __device__ __forceinline__ v8bf cvt8(f32x4 a, f32x4 b) {
  v8bf r;
  r[0]=(__bf16)a[0]; r[1]=(__bf16)a[1]; r[2]=(__bf16)a[2]; r[3]=(__bf16)a[3];
  r[4]=(__bf16)b[0]; r[5]=(__bf16)b[1]; r[6]=(__bf16)b[2]; r[7]=(__bf16)b[3];
  return r;
}
static __device__ __forceinline__ v8f wmma_bf16(v16bf a, v16bf b, v8f c) {
  // D = A(16x32 bf16) * B(32x16 bf16) + C(16x16 f32)
  return __builtin_amdgcn_wmma_f32_16x16x32_bf16(false, a, false, b, (short)0, c, false, false);
}

// ---------------------------------------------------------------------------
// Kernel 1: convert Wx,Wh -> transposed bf16 (WxT[h][k] = Wx[k][h]), h0 -> bf16,
// and reset the grid-barrier counter (deterministic per call).
// ---------------------------------------------------------------------------
__global__ __launch_bounds__(256) void prep_kernel(
    const float* __restrict__ Wx, const float* __restrict__ Wh,
    const float* __restrict__ h0, __bf16* __restrict__ WxT,
    __bf16* __restrict__ WhT, __bf16* __restrict__ h0b,
    unsigned* __restrict__ ctr) {
  int idx = blockIdx.x * 256 + threadIdx.x;
  if (idx == 0) *ctr = 0u;
  if (idx < DD * HH) {
    int h = idx >> 10, k = idx & 1023;
    WxT[idx] = (__bf16)Wx[(size_t)k * HH + h];
    WhT[idx] = (__bf16)Wh[(size_t)k * HH + h];
  }
  if (idx < NB * HH) h0b[idx] = (__bf16)h0[idx];
}

// ---------------------------------------------------------------------------
// Kernel 2: xw = x @ Wx + b, written into d_out (phase 2 overwrites in place).
// Block tile 128x128, 8 waves (4 along M x 2 along N), wave tile 32x64.
// Grid: x = N-tiles (8) fastest so blocks sharing an x-strip co-run (L2 reuse);
// y = M-tiles (512).  A staged via LDS (fp32->bf16 convert), B staged via
// async global->LDS copies when available.
// ---------------------------------------------------------------------------
__global__ __launch_bounds__(256) void gemm_xw_kernel(
    const float* __restrict__ x, const __bf16* __restrict__ WxT,
    const float* __restrict__ bias, float* __restrict__ out) {
  __shared__ alignas(16) __bf16 As[128][40];   // M x K, padded
  __shared__ alignas(16) __bf16 Bs[128][40];   // Ncols x K (B transposed), padded
  const int tid  = threadIdx.x;
  const int bn   = blockIdx.x;                 // 8  N-tiles (fastest)
  const int bm   = blockIdx.y;                 // 512 M-tiles
  const int wave = tid >> 5, lane = tid & 31;
  const int wm   = wave & 3, wn = wave >> 2;
  const int l16  = lane & 15, lhi = lane >> 4;

  v8f acc[2][4];
  #pragma unroll
  for (int i = 0; i < 2; i++)
    #pragma unroll
    for (int j = 0; j < 4; j++) {
      v8f z = {0.f,0.f,0.f,0.f,0.f,0.f,0.f,0.f};
      acc[i][j] = z;
    }

  // A tile loaders: thread -> (row = tid>>1, 16-col segment = (tid&1)*16)
  const int ar = tid >> 1;
  const int ac = (tid & 1) * 16;
  const float* aptr = x + (size_t)(bm * 128 + ar) * DD + ac;
  // B tile loaders: thread -> chunks tid and tid+256 of 512 (row=c>>2, k8=(c&3)*8)
  const int bhr = tid >> 2;
  const int bk8 = (tid & 3) * 8;
  const __bf16* bp0 = WxT + (size_t)(bn * 128 + bhr) * DD + bk8;
  const __bf16* bp1 = bp0 + (size_t)64 * DD;

  for (int k0 = 0; k0 < DD; k0 += 32) {
    f32x4 f0 = *(const f32x4*)(aptr + k0 + 0);
    f32x4 f1 = *(const f32x4*)(aptr + k0 + 4);
    f32x4 f2 = *(const f32x4*)(aptr + k0 + 8);
    f32x4 f3 = *(const f32x4*)(aptr + k0 + 12);
    *(v8bf*)&As[ar][ac]     = cvt8(f0, f1);
    *(v8bf*)&As[ar][ac + 8] = cvt8(f2, f3);
#if USE_ASYNC_LDS
    __builtin_amdgcn_global_load_async_to_lds_b128(
        (gas1_b128)(bp0 + k0), (las3_b128)&Bs[bhr][bk8], 0, 0);
    __builtin_amdgcn_global_load_async_to_lds_b128(
        (gas1_b128)(bp1 + k0), (las3_b128)&Bs[64 + bhr][bk8], 0, 0);
#else
    *(v8bf*)&Bs[bhr][bk8]      = *(const v8bf*)(bp0 + k0);
    *(v8bf*)&Bs[64 + bhr][bk8] = *(const v8bf*)(bp1 + k0);
#endif
    if (k0 + 32 < DD) __builtin_prefetch(aptr + k0 + 32, 0, 1);
#if USE_ASYNC_LDS
    __builtin_amdgcn_s_wait_asynccnt(0);
#endif
    __syncthreads();

    v16bf af[2];
    #pragma unroll
    for (int fm = 0; fm < 2; fm++) {
      const __bf16* p = &As[wm * 32 + fm * 16 + l16][lhi * 8];
      af[fm] = cat16(*(const v8bf*)p, *(const v8bf*)(p + 16));
    }
    #pragma unroll
    for (int fn = 0; fn < 4; fn++) {
      const __bf16* p = &Bs[wn * 64 + fn * 16 + l16][lhi * 8];
      v16bf bf = cat16(*(const v8bf*)p, *(const v8bf*)(p + 16));
      acc[0][fn] = wmma_bf16(af[0], bf, acc[0][fn]);
      acc[1][fn] = wmma_bf16(af[1], bf, acc[1][fn]);
    }
    __syncthreads();
  }

  #pragma unroll
  for (int fn = 0; fn < 4; fn++) {
    int col = bn * 128 + wn * 64 + fn * 16 + l16;
    float bv = bias[col];
    #pragma unroll
    for (int fm = 0; fm < 2; fm++) {
      int row0 = bm * 128 + wm * 32 + fm * 16 + lhi * 8;
      #pragma unroll
      for (int i = 0; i < 8; i++)
        out[(size_t)(row0 + i) * HH + col] = acc[fm][fn][i] + bv;
    }
  }
}

// ---------------------------------------------------------------------------
// Kernel 3: persistent scan, latency-optimized. 16 blocks cover the 128x1024
// hidden state: block tile 64(batch) x 128(H), 8 waves (2x4), wave tile 32x32.
// No LDS, no intra-step barriers: WMMA fragments are loaded directly from
// L2/L0-resident WhT and hbuf as per-lane b128 loads matching the ISA
// fragment layouts. One grid-wide barrier per timestep.
// ---------------------------------------------------------------------------
__global__ __launch_bounds__(256) void rnn_scan_kernel(
    const __bf16* __restrict__ WhT, const __bf16* __restrict__ h0b,
    __bf16* __restrict__ hbuf, float* __restrict__ out,
    unsigned* __restrict__ ctr) {
  const int tid  = threadIdx.x;
  const int bm   = blockIdx.x & 1;     // 2 tiles along batch
  const int bn   = blockIdx.x >> 1;    // 8 tiles along H
  const int wave = tid >> 5, lane = tid & 31;
  const int wm   = wave & 1, wn = wave >> 1;
  const int l16  = lane & 15, lhi = lane >> 4;

  const int arow0 = bm * 64 + wm * 32 + l16;    // fm adds +16
  const int bcol0 = bn * 128 + wn * 32 + l16;   // fn adds +16
  // B fragment base pointers (lane holds column bcol, K = k0 + lhi*8 (+16))
  const __bf16* w0 = WhT + (size_t)bcol0 * HH + lhi * 8;
  const __bf16* w1 = WhT + (size_t)(bcol0 + 16) * HH + lhi * 8;

  for (int t = 0; t < TT; t++) {
    const __bf16* hp = (t == 0) ? h0b : (hbuf + (size_t)((t - 1) & 1) * (NB * HH));
    // A fragment base pointers (lane holds row arow, K = k0 + lhi*8 (+16))
    const __bf16* a0p = hp + (size_t)arow0 * HH + lhi * 8;
    const __bf16* a1p = hp + (size_t)(arow0 + 16) * HH + lhi * 8;

    v8f acc[2][2];
    #pragma unroll
    for (int i = 0; i < 2; i++)
      #pragma unroll
      for (int j = 0; j < 2; j++) {
        v8f z = {0.f,0.f,0.f,0.f,0.f,0.f,0.f,0.f};
        acc[i][j] = z;
      }

    #pragma unroll 4
    for (int k0 = 0; k0 < HH; k0 += 32) {
      v16bf a0 = cat16(*(const v8bf*)(a0p + k0), *(const v8bf*)(a0p + k0 + 16));
      v16bf a1 = cat16(*(const v8bf*)(a1p + k0), *(const v8bf*)(a1p + k0 + 16));
      v16bf b0 = cat16(*(const v8bf*)(w0 + k0),  *(const v8bf*)(w0 + k0 + 16));
      v16bf b1 = cat16(*(const v8bf*)(w1 + k0),  *(const v8bf*)(w1 + k0 + 16));
      acc[0][0] = wmma_bf16(a0, b0, acc[0][0]);
      acc[1][0] = wmma_bf16(a1, b0, acc[1][0]);
      acc[0][1] = wmma_bf16(a0, b1, acc[0][1]);
      acc[1][1] = wmma_bf16(a1, b1, acc[1][1]);
    }

    __bf16* hn = hbuf + (size_t)(t & 1) * (NB * HH);
    #pragma unroll
    for (int fn = 0; fn < 2; fn++) {
      int col = bn * 128 + wn * 32 + fn * 16 + l16;
      #pragma unroll
      for (int fm = 0; fm < 2; fm++) {
        int n0 = bm * 64 + wm * 32 + fm * 16 + lhi * 8;
        #pragma unroll
        for (int i = 0; i < 8; i++) {
          int n = n0 + i;
          size_t idx = ((size_t)n * TT + t) * HH + col;   // (N,T,H) layout
          float v = tanhf(acc[fm][fn][i] + out[idx]);     // xw_t read, h_t written
          out[idx] = v;
          hn[(size_t)n * HH + col] = (__bf16)v;
        }
      }
    }

    // grid-wide barrier between timesteps
    __threadfence();
    __syncthreads();
    if (tid == 0) {
      __hip_atomic_fetch_add(ctr, 1u, __ATOMIC_RELEASE, __HIP_MEMORY_SCOPE_AGENT);
      unsigned target = (unsigned)(t + 1) * (unsigned)NBLK2;
      while (__hip_atomic_load(ctr, __ATOMIC_ACQUIRE, __HIP_MEMORY_SCOPE_AGENT) < target)
        __builtin_amdgcn_s_sleep(1);
    }
    __syncthreads();
  }
}

extern "C" void kernel_launch(void* const* d_in, const int* in_sizes, int n_in,
                              void* d_out, int out_size, void* d_ws, size_t ws_size,
                              hipStream_t stream) {
  (void)in_sizes; (void)n_in; (void)out_size; (void)ws_size;
  const float* x  = (const float*)d_in[0];
  const float* h0 = (const float*)d_in[1];
  const float* Wx = (const float*)d_in[2];
  const float* Wh = (const float*)d_in[3];
  const float* b  = (const float*)d_in[4];
  float* out = (float*)d_out;

  char* ws = (char*)d_ws;
  __bf16*   WxT  = (__bf16*)(ws);                                     // 2 MB
  __bf16*   WhT  = (__bf16*)(ws + (size_t)(2u << 20));                // 2 MB
  __bf16*   h0b  = (__bf16*)(ws + (size_t)(4u << 20));                // 256 KB
  __bf16*   hbuf = (__bf16*)(ws + (size_t)(4u << 20) + (256u << 10)); // 512 KB (double buffer)
  unsigned* ctr  = (unsigned*)(ws + (size_t)(4u << 20) + (768u << 10));

  prep_kernel<<<dim3((DD * HH + 255) / 256), dim3(256), 0, stream>>>(
      Wx, Wh, h0, WxT, WhT, h0b, ctr);
  // N-tiles fastest (x-dim) so the 8 blocks sharing one x-strip co-run.
  gemm_xw_kernel<<<dim3(HH / 128, (NB * TT) / 128), dim3(256), 0, stream>>>(
      x, WxT, b, out);
  rnn_scan_kernel<<<dim3(NBLK2), dim3(256), 0, stream>>>(
      WhT, h0b, hbuf, out, ctr);
}
